// EdgeUpdateLayer_75866302317034
// MI455X (gfx1250) — matrix operand order repaired and checked by reading, not compiled, not run on hardware
//
#include <hip/hip_runtime.h>

typedef __attribute__((ext_vector_type(16))) _Float16 v16h;
typedef __attribute__((ext_vector_type(8)))  _Float16 v8h;
typedef __attribute__((ext_vector_type(4)))  _Float16 v4h;
typedef __attribute__((ext_vector_type(8)))  float    v8f;

#define LDS_STRIDE 200   // 192 halves + 8 pad (avoids LDS bank conflicts)

// ---------------------------------------------------------------------------
// Prep: transpose + convert weights to f16, n-major (Wt[n][k] = W[k][n]).
// W1: [192][192] fp32 -> Wt1: [192][192] f16 ; W2: [192][64] fp32 -> Wt2: [64][192] f16
// ---------------------------------------------------------------------------
__global__ void prep_weights(const float* __restrict__ W1,
                             const float* __restrict__ W2,
                             _Float16* __restrict__ Wt1,
                             _Float16* __restrict__ Wt2) {
  int idx = blockIdx.x * blockDim.x + threadIdx.x;
  if (idx < 192 * 192) {
    int n = idx / 192, k = idx - n * 192;
    Wt1[idx] = (_Float16)W1[k * 192 + n];
  }
  int idx2 = idx - 192 * 192;
  if (idx2 >= 0 && idx2 < 64 * 192) {
    int n = idx2 / 192, k = idx2 - n * 192;
    Wt2[idx2] = (_Float16)W2[k * 64 + n];
  }
}

// ---------------------------------------------------------------------------
// Async global->LDS 16B copy (CDNA5, ASYNCcnt-tracked). GVS addressing:
// saddr = SGPR-pair base, vaddr = 32-bit byte offset, vdst = LDS byte address.
// Generic pointers to LDS carry the LDS offset in their low 32 bits (ISA 10.2).
// ---------------------------------------------------------------------------
__device__ __forceinline__ void async_copy_b128(const _Float16* gbase,
                                                unsigned byte_off,
                                                _Float16* lds_dst) {
#if defined(__gfx1250__)
  unsigned lds_addr = (unsigned)(unsigned long long)lds_dst;
  asm volatile("global_load_async_to_lds_b128 %0, %1, %2"
               :: "v"(lds_addr), "v"(byte_off), "s"(gbase)
               : "memory");
#else
  *(v8h*)lds_dst = *(const v8h*)((const char*)gbase + byte_off);
#endif
}

__device__ __forceinline__ void wait_async_copies() {
#if defined(__gfx1250__)
#if __has_builtin(__builtin_amdgcn_s_wait_asynccnt)
  __builtin_amdgcn_s_wait_asynccnt(0);
#else
  asm volatile("s_wait_asynccnt 0x0" ::: "memory");
#endif
#endif
}

// ---------------------------------------------------------------------------
// Fragment loaders (wave32 WMMA layouts, CDNA5 ISA 7.12.2)
// ---------------------------------------------------------------------------
// A fragment, 16x32 f16. Row M = lane&15. K halves per lane:
//   h=0..7  -> K = kb + h        (kb = 8 if lane>=16 else 0)
//   h=8..15 -> K = 16 + kb + (h-8)
__device__ __forceinline__ v16h load_a16(const _Float16* base, int lane, int k0) {
  int m  = lane & 15;
  int kb = (lane & 16) ? 8 : 0;
  const _Float16* p = base + m * LDS_STRIDE + k0 + kb;
  v8h x0 = *(const v8h*)(p);
  v8h x1 = *(const v8h*)(p + 16);
  v16h r;
#pragma unroll
  for (int t = 0; t < 8; ++t) { r[t] = x0[t]; r[t + 8] = x1[t]; }
  return r;
}

// B fragment, 32x16 f16, read from n-major transposed weights Wt[n][k].
// Col N = lane&15 ; h=0..7 -> K = k0 + kb + h, h=8..15 -> K = k0 + 16 + kb + (h-8) (kb = 8 if lane>=16).
__device__ __forceinline__ v16h load_b16(const _Float16* base, int lane, int k0) {
  int n  = lane & 15;
  int kb = (lane & 16) ? 8 : 0;
  const _Float16* p = base + n * LDS_STRIDE + k0 + kb;
  v8h x0 = *(const v8h*)(p);
  v8h x1 = *(const v8h*)(p + 16);
  v16h r;
#pragma unroll
  for (int t = 0; t < 8; ++t) { r[t] = x0[t]; r[t + 8] = x1[t]; }
  return r;
}

__device__ __forceinline__ v4h cvt4(float4 a) {
  v4h h;
  h[0] = (_Float16)a.x; h[1] = (_Float16)a.y;
  h[2] = (_Float16)a.z; h[3] = (_Float16)a.w;
  return h;
}

// ---------------------------------------------------------------------------
// Fused edge-update kernel: one block per (b,i): 128 rows (j) x 192 features.
// ---------------------------------------------------------------------------
__global__ __launch_bounds__(256, 1)
void edge_update_kernel(const float* __restrict__ node,    // [16,128,64]
                        const float* __restrict__ edge,    // [16,128,128,64]
                        const _Float16* __restrict__ Wt1,  // [192,192] n-major f16
                        const float* __restrict__ b1,
                        const float* __restrict__ g1,
                        const float* __restrict__ be1,
                        const _Float16* __restrict__ Wt2,  // [64,192] n-major f16
                        const float* __restrict__ b2,
                        const float* __restrict__ g2,
                        const float* __restrict__ be2,
                        float* __restrict__ out) {
  extern __shared__ _Float16 smem[];
  _Float16* Xs  = smem;                        // 128 x 200  (X tile, f16)
  _Float16* Hs  = smem + 128 * LDS_STRIDE;     // 128 x 200  (H tile, f16)
  _Float16* Ws1 = smem + 256 * LDS_STRIDE;     // 192 x 200  (W1^T)
  _Float16* Ws2 = smem + 448 * LDS_STRIDE;     //  64 x 200  (W2^T)

  const int tid = threadIdx.x;
  const int blk = blockIdx.x;                  // = b*128 + i
  const int b   = blk >> 7;
  const int i   = blk & 127;

  const float* nodeb = node + (size_t)b * 8192;   // node[b,:,:]
  const float* nfi   = nodeb + i * 64;            // node[b,i,:]
  const float* edgeb = edge + (size_t)blk * 8192; // edge[b,i,:,:]

  // ---- stage transposed weights: async DMA f16 global->LDS (ASYNCcnt) ----
  for (int q = tid; q < 4608; q += 256) {            // 192*192/8 chunks
    int n = q / 24, k = (q - n * 24) * 8;
    async_copy_b128(Wt1, (unsigned)q * 16u, Ws1 + n * LDS_STRIDE + k);
  }
  for (int q = tid; q < 1536; q += 256) {            // 64*192/8 chunks
    int n = q / 24, k = (q - n * 24) * 8;
    async_copy_b128(Wt2, (unsigned)q * 16u, Ws2 + n * LDS_STRIDE + k);
  }

  // ---- stage X tile: row j = [node[b,j] | node[b,i] | edge[b,i,j]] ----
  // float4 loads (coalesced), fp32->f16 convert, 8B LDS stores.
  // Overlaps with the async weight DMA above.
  for (int q = tid; q < 2048; q += 256) {
    int j  = q >> 4;
    int c4 = (q & 15) << 2;
    float4 a = *(const float4*)(nodeb + (q << 2));
    float4 c = *(const float4*)(nfi + c4);
    float4 e = *(const float4*)(edgeb + (q << 2));
    _Float16* row = Xs + j * LDS_STRIDE;
    *(v4h*)(row + c4)       = cvt4(a);
    *(v4h*)(row + 64 + c4)  = cvt4(c);
    *(v4h*)(row + 128 + c4) = cvt4(e);
  }

  wait_async_copies();
  __syncthreads();

  const int lane = tid & 31;
  const int wave = tid >> 5;
  const int row0 = wave * 16;                  // this wave's 16 rows (j)
  const int N    = lane & 15;                  // C-tile column within tile
  const int mb   = (lane & 16) ? 8 : 0;        // C-tile row base

  // =================== GEMM1: [16x192] = [16x192]x[192x192] ===============
  v16h afrag[6];
#pragma unroll
  for (int ks = 0; ks < 6; ++ks)
    afrag[ks] = load_a16(Xs + row0 * LDS_STRIDE, lane, ks * 32);

  v8f acc1[12];
#pragma unroll
  for (int t = 0; t < 12; ++t) {
    v8f c = {};
#pragma unroll
    for (int ks = 0; ks < 6; ++ks) {
      v16h bfrag = load_b16(Ws1 + (t * 16) * LDS_STRIDE, lane, ks * 32);
      c = __builtin_amdgcn_wmma_f32_16x16x32_f16(false, afrag[ks], false, bfrag,
                                                 (short)0, c, false, false);
    }
    acc1[t] = c;
  }

  // ---- epilogue 1: +b1, LayerNorm(192), ReLU, store f16 H to LDS ----
  float s[8], ss[8];
#pragma unroll
  for (int f = 0; f < 8; ++f) { s[f] = 0.f; ss[f] = 0.f; }
#pragma unroll
  for (int t = 0; t < 12; ++t) {
    float bias = b1[t * 16 + N];
#pragma unroll
    for (int f = 0; f < 8; ++f) {
      float v = acc1[t][f] + bias;
      acc1[t][f] = v;
      s[f] += v;
      ss[f] += v * v;
    }
  }
#pragma unroll
  for (int m = 1; m <= 8; m <<= 1) {          // reduce across 16-lane half
#pragma unroll
    for (int f = 0; f < 8; ++f) {
      s[f]  += __shfl_xor(s[f],  m, 32);
      ss[f] += __shfl_xor(ss[f], m, 32);
    }
  }
  float mean[8], rstd[8];
#pragma unroll
  for (int f = 0; f < 8; ++f) {
    float mu  = s[f] * (1.0f / 192.0f);
    float var = ss[f] * (1.0f / 192.0f) - mu * mu;
    mean[f] = mu;
    rstd[f] = __frsqrt_rn(var + 1e-5f);
  }
#pragma unroll
  for (int t = 0; t < 12; ++t) {
    int col = t * 16 + N;
    float gg = g1[col], bb = be1[col];
#pragma unroll
    for (int f = 0; f < 8; ++f) {
      float h = (acc1[t][f] - mean[f]) * rstd[f] * gg + bb;
      h = fmaxf(h, 0.0f);
      Hs[(row0 + mb + f) * LDS_STRIDE + col] = (_Float16)h;
    }
  }
  __syncthreads();

  // =================== GEMM2: [16x64] = [16x192]x[192x64] =================
  v16h hfrag[6];
#pragma unroll
  for (int ks = 0; ks < 6; ++ks)
    hfrag[ks] = load_a16(Hs + row0 * LDS_STRIDE, lane, ks * 32);

  v8f acc2[4];
#pragma unroll
  for (int t = 0; t < 4; ++t) {
    v8f c = {};
#pragma unroll
    for (int ks = 0; ks < 6; ++ks) {
      v16h bfrag = load_b16(Ws2 + (t * 16) * LDS_STRIDE, lane, ks * 32);
      c = __builtin_amdgcn_wmma_f32_16x16x32_f16(false, hfrag[ks], false, bfrag,
                                                 (short)0, c, false, false);
    }
    acc2[t] = c;
  }

  // ---- epilogue 2: +b2, +edge residual (fp32), LayerNorm(64), store ----
  float s2[8], ss2[8];
#pragma unroll
  for (int f = 0; f < 8; ++f) { s2[f] = 0.f; ss2[f] = 0.f; }
#pragma unroll
  for (int t = 0; t < 4; ++t) {
    int col = t * 16 + N;
    float bias = b2[col];
#pragma unroll
    for (int f = 0; f < 8; ++f) {
      int r = row0 + mb + f;
      float v = acc2[t][f] + bias + edgeb[r * 64 + col];
      acc2[t][f] = v;
      s2[f] += v;
      ss2[f] += v * v;
    }
  }
#pragma unroll
  for (int m = 1; m <= 8; m <<= 1) {
#pragma unroll
    for (int f = 0; f < 8; ++f) {
      s2[f]  += __shfl_xor(s2[f],  m, 32);
      ss2[f] += __shfl_xor(ss2[f], m, 32);
    }
  }
#pragma unroll
  for (int f = 0; f < 8; ++f) {
    float mu  = s2[f] * (1.0f / 64.0f);
    float var = ss2[f] * (1.0f / 64.0f) - mu * mu;
    s2[f]  = mu;
    ss2[f] = __frsqrt_rn(var + 1e-5f);
  }
  float* outb = out + (size_t)blk * 8192;
#pragma unroll
  for (int t = 0; t < 4; ++t) {
    int col = t * 16 + N;
    float gg = g2[col], bb = be2[col];
#pragma unroll
    for (int f = 0; f < 8; ++f) {
      int r = row0 + mb + f;
      outb[r * 64 + col] = (acc2[t][f] - s2[f]) * ss2[f] * gg + bb;
    }
  }
}

// ---------------------------------------------------------------------------
extern "C" void kernel_launch(void* const* d_in, const int* in_sizes, int n_in,
                              void* d_out, int out_size, void* d_ws, size_t ws_size,
                              hipStream_t stream) {
  (void)in_sizes; (void)n_in; (void)out_size; (void)ws_size;
  const float* node = (const float*)d_in[0];
  const float* edge = (const float*)d_in[1];
  const float* W1   = (const float*)d_in[2];
  const float* b1   = (const float*)d_in[3];
  const float* g1   = (const float*)d_in[4];
  const float* be1  = (const float*)d_in[5];
  const float* W2   = (const float*)d_in[6];
  const float* b2   = (const float*)d_in[7];
  const float* g2   = (const float*)d_in[8];
  const float* be2  = (const float*)d_in[9];
  float* out = (float*)d_out;

  _Float16* Wt1 = (_Float16*)d_ws;                                         // 192*192 f16
  _Float16* Wt2 = (_Float16*)((char*)d_ws + 192 * 192 * sizeof(_Float16)); // 64*192 f16

  prep_weights<<<192, 256, 0, stream>>>(W1, W2, Wt1, Wt2);

  // LDS: (128 + 128 + 192 + 64) rows * 200 halves * 2 B = 204,800 B (< 320 KB/WGP)
  size_t lds_bytes = (size_t)512 * LDS_STRIDE * sizeof(_Float16);
  edge_update_kernel<<<2048, 256, lds_bytes, stream>>>(
      node, edge, Wt1, b1, g1, be1, Wt2, b2, g2, be2, out);
}
